// PatchDiffusion3D_13675175870635
// MI455X (gfx1250) — matrix operand order repaired and checked by reading, not compiled, not run on hardware
//
#include <hip/hip_runtime.h>
#include <math.h>

// ---------------- problem constants (match reference) ----------------
#define NBATCH 2
#define DD 32
#define HH 256
#define WW 256
#define CC 64
#define NPER 75000
#define MPTS (NBATCH*NPER)            // 150000 input points
#define NVOX (NBATCH*DD*HH*WW)        // 4194304 dense voxels
#define NBLK (NVOX/256)               // 16384 scan blocks
#define VMAX 900000                   // upper bound on unique expanded voxels (6*M)
#define NTAP 27
#define WHALVES 4096                  // packed weight halves per tap (2*4*32*16)

typedef __attribute__((ext_vector_type(16))) _Float16 v16h;
typedef __attribute__((ext_vector_type(8)))  _Float16 v8h;
typedef __attribute__((ext_vector_type(8)))  float    v8f;

// meta[] layout: 0:V 1:NRAW 2..3:diffcnt[b] 4..5:thrbin[b] 6..7:rem[b]
//                8..9:K[b] 10..11:base[b] 12..13:emitHi[b] 14..15:emitEq[b]

__device__ __forceinline__ int voxidx(int b,int d,int h,int w){
  return ((b*WW + w)*HH + h)*DD + d;        // == key b*SXYZ + w*SYZ + h*SZ + d
}
__device__ __forceinline__ void voxdec(int v,int&b,int&d,int&h,int&w){
  d = v % DD; h = (v/DD) % HH; w = (v/(DD*HH)) % WW; b = v/(DD*HH*WW);
}
__device__ __forceinline__ int iclamp(int x,int lo,int hi){ return x<lo?lo:(x>hi?hi:x); }

// diffusion expansion, copies 0..5 (copy 0 and 5 are the original coord)
__device__ __forceinline__ void expand(int cp,int&d,int&h,int&w){
  switch(cp){
    case 1: { int dw = (2*w > WW) ? 1 : -1; w = iclamp(w+dw,0,WW-1); } break;
    case 2: { int dh = (2*h > HH) ? 1 : -1; h = iclamp(h+dh,0,HH-1); } break;
    case 3: d = iclamp(d-1,0,DD-1); break;
    case 4: d = iclamp(d+1,0,DD-1); break;
    default: break;
  }
}

// ---------------- generic fills ----------------
__global__ void pd_fill_u32(unsigned* p, long n, unsigned v){
  long i = (long)blockIdx.x*256 + threadIdx.x, st = (long)gridDim.x*256;
  for(; i<n; i+=st) p[i]=v;
}

// ---------------- f16 feature staging (with zero pad row) ----------------
__global__ void pd_feat16(const float* __restrict__ f, _Float16* __restrict__ o){
  long n = (long)(MPTS+1)*CC;
  long i = (long)blockIdx.x*256 + threadIdx.x;
  if(i<n) o[i] = (i < (long)MPTS*CC) ? (_Float16)f[i] : (_Float16)0.f;
}

// ---------------- weight repack into exact WMMA B-lane layout ----------------
// wpack[k][kc][t][lane][q] = W[k][32*kc + 8*(lane>>4) + (q&7) + 16*(q>>3)][16*t + (lane&15)]
__global__ void pd_packw(const float* __restrict__ w, _Float16* __restrict__ o){
  int i = blockIdx.x*256 + threadIdx.x;
  if(i >= NTAP*WHALVES) return;
  int q=i&15, lane=(i>>4)&31, t=(i>>9)&3, kc=(i>>11)&1, k=i>>12;
  int kk = 32*kc + 8*(lane>>4) + (q&7) + 16*(q>>3);
  int nn = 16*t + (lane&15);
  o[i] = (_Float16)w[(long)k*CC*CC + (long)kk*CC + nn];
}

// ---------------- dense point table + neighbor tables ----------------
__global__ void pd_scatter(const int* __restrict__ idxs, int* __restrict__ table){
  int i = blockIdx.x*256 + threadIdx.x; if(i>=MPTS) return;
  table[voxidx(idxs[4*i],idxs[4*i+1],idxs[4*i+2],idxs[4*i+3])] = i;
}
__global__ void pd_nei1(const int* __restrict__ idxs, const int* __restrict__ table,
                        int* __restrict__ nei){
  int i = blockIdx.x*256 + threadIdx.x; if(i>=MPTS) return;
  int b=idxs[4*i], d=idxs[4*i+1], h=idxs[4*i+2], w=idxs[4*i+3];
  #pragma unroll
  for(int k=0;k<NTAP;++k){
    int dz=k/9-1, dy=(k/3)%3-1, dx=k%3-1;
    int z=d+dz, y=h+dy, x=w+dx, r=-1;
    if(z>=0&&z<DD&&y>=0&&y<HH&&x>=0&&x<WW) r = table[voxidx(b,z,y,x)];
    nei[(long)i*NTAP+k] = r;
  }
}

// ---------------- WMMA submanifold conv + LayerNorm + GELU ----------------
// B tiles staged in LDS (double buffered, block-cooperative), A gather
// software-pipelined one tap ahead.
__global__ __launch_bounds__(256) void pd_subm_conv(
    const _Float16* __restrict__ fin,   // [(rows+1)][64] f16, row `zrow` is zeros
    const int*      __restrict__ nei,   // [rows][27]
    const _Float16* __restrict__ wpack, // [27][2][4][32][16] f16
    const float*    __restrict__ gam, const float* __restrict__ bet,
    _Float16* __restrict__ out16, float* __restrict__ out32, float* __restrict__ act,
    int nrowsMax, const int* __restrict__ nrowsDev, int zrow)
{
  __shared__ _Float16 wsh[2][WHALVES];       // 2 x 8KB double-buffered weights
  __shared__ float tile[8][16][CC+1];        // LN staging
  const int lane = threadIdx.x & 31;
  const int wave = threadIdx.x >> 5;
  const int half = lane >> 4;
  const int mr   = lane & 15;
  const int Mrows = nrowsDev ? *nrowsDev : nrowsMax;
  const int rowBase = (blockIdx.x*8 + wave)*16;
  const int myrow = rowBase + mr;

  v8f zero;
  #pragma unroll
  for(int e=0;e<8;++e) zero[e]=0.0f;
  v8f acc[4] = {zero, zero, zero, zero};

  union V16 { v16h v; v8h h[2]; };

  // stage tap-0 weights: 256 threads x 32B = 8KB
  ((v16h*)wsh[0])[threadIdx.x] = ((const v16h*)wpack)[threadIdx.x];

  // A tiles for tap 0 (K=0..31 and K=32..63)
  V16 a0, a1;
  {
    int nb = (myrow < Mrows) ? nei[(long)myrow*NTAP + 0] : -1;
    const _Float16* rp = fin + (size_t)((nb<0)?zrow:nb)*CC;
    a0.h[0] = *(const v8h*)(rp +  8*half);
    a0.h[1] = *(const v8h*)(rp + 16 + 8*half);
    a1.h[0] = *(const v8h*)(rp + 32 + 8*half);
    a1.h[1] = *(const v8h*)(rp + 48 + 8*half);
  }
  __syncthreads();

  for(int k=0;k<NTAP;++k){
    const int buf = k & 1;
    // pipeline: stage next tap's weights into the other LDS buffer
    if(k+1 < NTAP)
      ((v16h*)wsh[buf^1])[threadIdx.x] =
          ((const v16h*)(wpack + (size_t)(k+1)*WHALVES))[threadIdx.x];
    // pipeline: gather next tap's A tiles into registers
    V16 na0, na1;
    if(k+1 < NTAP){
      int nb = (myrow < Mrows) ? nei[(long)myrow*NTAP + (k+1)] : -1;
      const _Float16* rp = fin + (size_t)((nb<0)?zrow:nb)*CC;
      na0.h[0] = *(const v8h*)(rp +  8*half);
      na0.h[1] = *(const v8h*)(rp + 16 + 8*half);
      na1.h[0] = *(const v8h*)(rp + 32 + 8*half);
      na1.h[1] = *(const v8h*)(rp + 48 + 8*half);
    }
    // compute with current weights from LDS (ds_load_b128 pairs)
    const _Float16* wb = wsh[buf];
    #pragma unroll
    for(int t=0;t<4;++t){
      V16 b0, b1;
      b0.v = *(const v16h*)(wb + ((0*4+t)*32 + lane)*16);
      b1.v = *(const v16h*)(wb + ((1*4+t)*32 + lane)*16);
      acc[t] = __builtin_amdgcn_wmma_f32_16x16x32_f16(false, a0.v, false, b0.v,
                                                      (short)0, acc[t], false, false);
      acc[t] = __builtin_amdgcn_wmma_f32_16x16x32_f16(false, a1.v, false, b1.v,
                                                      (short)0, acc[t], false, false);
    }
    if(k+1 < NTAP){ a0 = na0; a1 = na1; }
    __syncthreads();   // next buffer fully written before it is read
  }

  // D layout: lane l, acc t, elem j -> out[M = j+8*(l>>4)][N = 16t + (l&15)]
  #pragma unroll
  for(int t=0;t<4;++t)
    #pragma unroll
    for(int j=0;j<8;++j)
      tile[wave][j + 8*half][16*t + mr] = acc[t][j];
  __syncthreads();

  if(half==0 && myrow < Mrows){
    float s=0.f;
    #pragma unroll
    for(int n=0;n<CC;++n) s += tile[wave][mr][n];
    float mean = s*(1.0f/CC), vv=0.f;
    #pragma unroll
    for(int n=0;n<CC;++n){ float dd = tile[wave][mr][n]-mean; vv += dd*dd; }
    float rstd = rsqrtf(vv*(1.0f/CC) + 1e-5f);
    float asum = 0.f;
    for(int n=0;n<CC;++n){
      float y = (tile[wave][mr][n]-mean)*rstd*gam[n] + bet[n];
      y = 0.5f*y*(1.0f + erff(y*0.70710678118f));   // exact GELU
      asum += y;
      if(out16) out16[(long)myrow*CC+n] = (_Float16)y;
      if(out32) out32[(long)myrow*CC+n] = y;
    }
    if(act) act[myrow] = asum*(1.0f/CC);
  }
}

// ---------------- expansion marking ----------------
__global__ void pd_mark(const int* __restrict__ idxs, unsigned* __restrict__ active){
  int t = blockIdx.x*256 + threadIdx.x; if(t>=MPTS*6) return;
  int i=t/6, cp=t%6;
  int b=idxs[4*i], d=idxs[4*i+1], h=idxs[4*i+2], w=idxs[4*i+3];
  expand(cp,d,h,w);
  active[voxidx(b,d,h,w)] = 1u;
}
__global__ void pd_rawflag(const unsigned* __restrict__ active,
                           const int* __restrict__ table, unsigned* __restrict__ rawf){
  int v = blockIdx.x*256 + threadIdx.x; if(v>=NVOX) return;
  rawf[v] = (active[v] && table[v]>=0) ? 1u : 0u;
}

// ---------------- deterministic exclusive scan (ids in sorted-key order) ----------------
__global__ void pd_scan_blocks(const unsigned* __restrict__ in, unsigned* __restrict__ out,
                               unsigned* __restrict__ bsums){
  __shared__ unsigned sh[256];
  int i = blockIdx.x*256 + threadIdx.x;
  unsigned v = (i<NVOX) ? in[i] : 0u;
  sh[threadIdx.x] = v; __syncthreads();
  for(int off=1; off<256; off<<=1){
    unsigned t = (threadIdx.x>=off) ? sh[threadIdx.x-off] : 0u;
    __syncthreads();
    sh[threadIdx.x] += t;
    __syncthreads();
  }
  if(i<NVOX) out[i] = sh[threadIdx.x] - v;   // exclusive
  if(threadIdx.x==255) bsums[blockIdx.x] = sh[255];
}
__global__ __launch_bounds__(1024) void pd_scan_partials(unsigned* __restrict__ bs,
                                                         unsigned* __restrict__ totalOut){
  __shared__ unsigned sh[1024];
  unsigned loc[16]; unsigned s=0;
  int base = threadIdx.x*16;
  #pragma unroll
  for(int j=0;j<16;++j){ unsigned v = bs[base+j]; loc[j]=s; s+=v; }
  sh[threadIdx.x]=s; __syncthreads();
  for(int off=1; off<1024; off<<=1){
    unsigned t=(threadIdx.x>=off)?sh[threadIdx.x-off]:0u;
    __syncthreads(); sh[threadIdx.x]+=t; __syncthreads();
  }
  unsigned excl = sh[threadIdx.x]-s;
  #pragma unroll
  for(int j=0;j<16;++j) bs[base+j] = excl + loc[j];
  if(threadIdx.x==1023 && totalOut) *totalOut = sh[1023];
}
__global__ void pd_scan_add(unsigned* __restrict__ pref, const unsigned* __restrict__ bs){
  int i = blockIdx.x*256 + threadIdx.x; if(i<NVOX) pref[i] += bs[i>>8];
}

// ---------------- id assignment ----------------
__global__ void pd_assign(const unsigned* __restrict__ active, const unsigned* __restrict__ prefA,
                          const int* __restrict__ table, int* __restrict__ uid,
                          int* __restrict__ id_vox, int* __restrict__ meta){
  int v = blockIdx.x*256 + threadIdx.x; if(v>=NVOX) return;
  if(active[v]){
    int id = (int)prefA[v];
    uid[v] = id; id_vox[id] = v;
    if(table[v] < 0){ int b,d,h,w; voxdec(v,b,d,h,w); atomicAdd(&meta[2+b],1); }
  } else uid[v] = -1;
}

// ---------------- segment-sum (scatter-mean numerator) ----------------
__global__ void pd_accum(const int* __restrict__ idxs, const _Float16* __restrict__ h16,
                         const int* __restrict__ uid, float* __restrict__ xsum,
                         unsigned* __restrict__ cnt){
  long t = (long)blockIdx.x*256 + threadIdx.x;
  if(t >= (long)MPTS*6*16) return;
  int i  = (int)(t/96);
  int r  = (int)(t%96);
  int cp = r/16, cg = r%16;
  int b=idxs[4*i], d=idxs[4*i+1], h=idxs[4*i+2], w=idxs[4*i+3];
  expand(cp,d,h,w);
  int id = uid[voxidx(b,d,h,w)];
  const _Float16* hp = h16 + (long)i*CC + cg*4;
  float* xp = xsum + (long)id*CC + cg*4;
  #pragma unroll
  for(int c=0;c<4;++c) atomicAdd(xp+c, (float)hp[c]);
  if(cg==0) atomicAdd(&cnt[id],1u);
}

// ---------------- scatter-mean + middle LayerNorm -> xm (f16) ----------------
__global__ void pd_meanln(const float* __restrict__ xsum, const unsigned* __restrict__ cnt,
                          const float* __restrict__ g, const float* __restrict__ b,
                          _Float16* __restrict__ xm, const int* __restrict__ meta){
  int id = blockIdx.x*256 + threadIdx.x; if(id >= meta[0]) return;
  float inv = 1.0f/(float)cnt[id];
  float x[CC]; float s=0.f;
  #pragma unroll
  for(int n=0;n<CC;++n){ x[n] = xsum[(long)id*CC+n]*inv; s += x[n]; }
  float mean = s*(1.0f/CC), vv=0.f;
  #pragma unroll
  for(int n=0;n<CC;++n){ float d=x[n]-mean; vv+=d*d; }
  float rstd = rsqrtf(vv*(1.0f/CC)+1e-5f);
  #pragma unroll
  for(int n=0;n<CC;++n) xm[(long)id*CC+n] = (_Float16)((x[n]-mean)*rstd*g[n]+b[n]);
}

// ---------------- second neighbor table over unique voxels ----------------
__global__ void pd_nei2(const int* __restrict__ id_vox, const int* __restrict__ uid,
                        int* __restrict__ nei, const int* __restrict__ meta){
  int id = blockIdx.x*256 + threadIdx.x; if(id >= meta[0]) return;
  int b,d,h,w; voxdec(id_vox[id],b,d,h,w);
  #pragma unroll
  for(int k=0;k<NTAP;++k){
    int dz=k/9-1, dy=(k/3)%3-1, dx=k%3-1;
    int z=d+dz, y=h+dy, x=w+dx, r=-1;
    if(z>=0&&z<DD&&y>=0&&y<HH&&x>=0&&x<WW) r = uid[voxidx(b,z,y,x)];
    nei[(long)id*NTAP+k] = r;
  }
}

// ---------------- per-batch histogram top-k selection ----------------
__device__ __forceinline__ unsigned fkey(float a){
  unsigned u = __float_as_uint(a);
  return (u & 0x80000000u) ? ~u : (u | 0x80000000u);   // monotone order key
}
__global__ void pd_hist(const int* __restrict__ id_vox, const int* __restrict__ table,
                        const float* __restrict__ act, unsigned* __restrict__ hist,
                        const int* __restrict__ meta){
  int id = blockIdx.x*256 + threadIdx.x; if(id >= meta[0]) return;
  int v = id_vox[id]; if(table[v]>=0) return;          // raw voxel -> not diff
  int b,d,h,w; voxdec(v,b,d,h,w);
  atomicAdd(&hist[b*4096 + (fkey(act[id])>>20)], 1u);
}
__global__ void pd_thresh(const unsigned* __restrict__ hist, int* __restrict__ meta){
  int b = threadIdx.x;
  if(b < NBATCH){
    int K = meta[2+b]/4;                                // int(diffcnt * 0.25)
    meta[8+b] = K;
    unsigned cum=0; int thr=0; int rem=K;
    for(int bin=4095; bin>=0; --bin){
      unsigned c = hist[b*4096+bin];
      if(cum + c >= (unsigned)K){ thr=bin; rem=K-(int)cum; break; }
      cum += c;
    }
    meta[4+b]=thr; meta[6+b]=rem;
  }
  __syncthreads();
  if(threadIdx.x==0){
    int nraw = meta[1];
    meta[10] = nraw;
    meta[11] = nraw + meta[8];
  }
}
__global__ void pd_emit(const int* __restrict__ id_vox, const int* __restrict__ table,
                        const unsigned* __restrict__ prefR, const float* __restrict__ h2,
                        const float* __restrict__ act, int* __restrict__ meta,
                        float* __restrict__ dout, long coordBase){
  int id = blockIdx.x*256 + threadIdx.x; if(id >= meta[0]) return;
  int v = id_vox[id];
  int b,d,h,w; voxdec(v,b,d,h,w);
  int row;
  if(table[v] >= 0){
    row = (int)prefR[v];                                // raw rows, unique-key order
  } else {
    int bin = (int)(fkey(act[id])>>20);
    int thr = meta[4+b];
    if(bin > thr){
      row = meta[10+b] + atomicAdd(&meta[12+b],1);
    } else if(bin == thr){
      int s = atomicAdd(&meta[14+b],1);
      if(s >= meta[6+b]) return;
      row = meta[10+b] + meta[8+b] - 1 - s;
    } else return;
  }
  for(int c=0;c<CC;++c) dout[(long)row*CC+c] = h2[(long)id*CC+c];
  dout[coordBase + (long)row*4 + 0] = (float)b;
  dout[coordBase + (long)row*4 + 1] = (float)d;
  dout[coordBase + (long)row*4 + 2] = (float)h;
  dout[coordBase + (long)row*4 + 3] = (float)w;
}

// =====================================================================
extern "C" void kernel_launch(void* const* d_in, const int* in_sizes, int n_in,
                              void* d_out, int out_size, void* d_ws, size_t ws_size,
                              hipStream_t stream){
  const float* features = (const float*)d_in[0];
  const int*   indices  = (const int*)  d_in[1];
  const float* W1 = (const float*)d_in[2];
  const float* g1 = (const float*)d_in[3];
  const float* b1 = (const float*)d_in[4];
  const float* gn = (const float*)d_in[5];
  const float* bn = (const float*)d_in[6];
  const float* W2 = (const float*)d_in[7];
  const float* g2 = (const float*)d_in[8];
  const float* b2 = (const float*)d_in[9];
  float* dout = (float*)d_out;

  // ------- workspace carve -------
  size_t off = 0;
  auto carve = [&](size_t bytes)->void*{
    void* p = (char*)d_ws + off; off += (bytes + 255) & ~(size_t)255; return p;
  };
  int*      table1  = (int*)     carve((size_t)NVOX*4);
  unsigned* active  = (unsigned*)carve((size_t)NVOX*4);
  unsigned* rawf    = (unsigned*)carve((size_t)NVOX*4);
  unsigned* prefA   = (unsigned*)carve((size_t)NVOX*4);
  unsigned* prefR   = (unsigned*)carve((size_t)NVOX*4);
  int*      uid     = (int*)     carve((size_t)NVOX*4);
  int*      id_vox  = (int*)     carve((size_t)VMAX*4);
  unsigned* bsums   = (unsigned*)carve((size_t)NBLK*4);
  int*      meta    = (int*)     carve(1024*4);
  int*      nei1    = (int*)     carve((size_t)MPTS*NTAP*4);
  int*      nei2    = (int*)     carve((size_t)VMAX*NTAP*4);
  _Float16* f16in   = (_Float16*)carve((size_t)(MPTS+1)*CC*2);
  _Float16* h16     = (_Float16*)carve((size_t)(MPTS+1)*CC*2);
  _Float16* wpack1  = (_Float16*)carve((size_t)NTAP*WHALVES*2);
  _Float16* wpack2  = (_Float16*)carve((size_t)NTAP*WHALVES*2);
  float*    xsum    = (float*)   carve((size_t)(VMAX+1)*CC*4);
  unsigned* cnt     = (unsigned*)carve((size_t)VMAX*4);
  _Float16* xm      = (_Float16*)carve((size_t)(VMAX+1)*CC*2);
  float*    h2      = (float*)   carve((size_t)VMAX*CC*4);
  float*    act     = (float*)   carve((size_t)VMAX*4);
  unsigned* hist    = (unsigned*)carve((size_t)NBATCH*4096*4);
  (void)ws_size; (void)n_in; (void)in_sizes;

  const long Nout = (long)out_size / (CC + 4);
  const long coordBase = Nout * CC;

  // ------- init -------
  pd_fill_u32<<<4096,256,0,stream>>>((unsigned*)table1, NVOX, 0xFFFFFFFFu);
  pd_fill_u32<<<4096,256,0,stream>>>(active, NVOX, 0u);
  pd_fill_u32<<<4096,256,0,stream>>>((unsigned*)meta, 1024, 0u);
  pd_fill_u32<<<4096,256,0,stream>>>((unsigned*)xsum, (long)(VMAX+1)*CC, 0u);
  pd_fill_u32<<<4096,256,0,stream>>>(cnt, VMAX, 0u);
  pd_fill_u32<<<4096,256,0,stream>>>((unsigned*)xm, (long)(VMAX+1)*CC/2, 0u);
  pd_fill_u32<<<4096,256,0,stream>>>((unsigned*)h16, (long)(MPTS+1)*CC/2, 0u);
  pd_fill_u32<<<32,256,0,stream>>>(hist, NBATCH*4096, 0u);

  // ------- staging -------
  pd_feat16<<<((MPTS+1)*CC+255)/256,256,0,stream>>>(features, f16in);
  pd_packw <<<(NTAP*WHALVES+255)/256,256,0,stream>>>(W1, wpack1);
  pd_packw <<<(NTAP*WHALVES+255)/256,256,0,stream>>>(W2, wpack2);

  // ------- plan: table + nei1 -------
  pd_scatter<<<(MPTS+255)/256,256,0,stream>>>(indices, table1);
  pd_nei1   <<<(MPTS+255)/256,256,0,stream>>>(indices, table1, nei1);

  // ------- conv1 (WMMA) + LN + GELU -> h16 -------
  pd_subm_conv<<<(MPTS+127)/128,256,0,stream>>>(f16in, nei1, wpack1, g1, b1,
                                                h16, nullptr, nullptr,
                                                MPTS, nullptr, MPTS);

  // ------- expansion, unique ids (deterministic, sorted-key order) -------
  pd_mark   <<<(MPTS*6+255)/256,256,0,stream>>>(indices, active);
  pd_rawflag<<<NBLK,256,0,stream>>>(active, table1, rawf);

  pd_scan_blocks  <<<NBLK,256,0,stream>>>(active, prefA, bsums);
  pd_scan_partials<<<1,1024,0,stream>>>(bsums, (unsigned*)&meta[0]);   // V
  pd_scan_add     <<<NBLK,256,0,stream>>>(prefA, bsums);

  pd_scan_blocks  <<<NBLK,256,0,stream>>>(rawf, prefR, bsums);
  pd_scan_partials<<<1,1024,0,stream>>>(bsums, (unsigned*)&meta[1]);   // NRAW
  pd_scan_add     <<<NBLK,256,0,stream>>>(prefR, bsums);

  pd_assign<<<NBLK,256,0,stream>>>(active, prefA, table1, uid, id_vox, meta);

  // ------- scatter-mean + middle LN -> xm -------
  pd_accum <<<(int)(((long)MPTS*6*16+255)/256),256,0,stream>>>(indices, h16, uid, xsum, cnt);
  pd_meanln<<<(VMAX+255)/256,256,0,stream>>>(xsum, cnt, gn, bn, xm, meta);

  // ------- nei2 + conv2 (WMMA) + LN + GELU -> h2, act -------
  pd_nei2<<<(VMAX+255)/256,256,0,stream>>>(id_vox, uid, nei2, meta);
  pd_subm_conv<<<(VMAX+127)/128,256,0,stream>>>(xm, nei2, wpack2, g2, b2,
                                                nullptr, h2, act,
                                                VMAX, meta, VMAX);

  // ------- per-batch top-k + output assembly -------
  pd_hist  <<<(VMAX+255)/256,256,0,stream>>>(id_vox, table1, act, hist, meta);
  pd_thresh<<<1,64,0,stream>>>(hist, meta);
  pd_emit  <<<(VMAX+255)/256,256,0,stream>>>(id_vox, table1, prefR, h2, act, meta,
                                             dout, coordBase);
}
